// VoxelMorph3d_39513699123258
// MI455X (gfx1250) — matrix operand order repaired and checked by reading, not compiled, not run on hardware
//
#include <hip/hip_runtime.h>
#include <hip/hip_bf16.h>

// ============================================================================
// VoxelMorph 3D UNet on MI455X (gfx1250, wave32)
//  - 3x3x3 convs = implicit GEMM on V_WMMA_F32_16X16X32_BF16 (bf16 in, f32 acc)
//  - K-step 64, 2 WMMAs per barrier pair.
//  - im2col loader: lane = voxel (coalesced 128B/wave loads), wave = 8 K-slots
//    with wave-uniform (ci,tap) bookkeeping in SGPRs (readfirstlane),
//    fully branchless halo (unconditional toff read + cndmask on offset).
//  - BN uses batch statistics (training mode) like the reference.
// ============================================================================

typedef __bf16 bf16_t;
typedef __attribute__((ext_vector_type(16))) __bf16 v16bf;
typedef __attribute__((ext_vector_type(8)))  __bf16 v8bf;
typedef __attribute__((ext_vector_type(8)))  float  v8f;

#define BN_EPS 1e-5f

// ---------------------------------------------------------------------------
__global__ void concat2_kernel(const float* __restrict__ a,
                               const float* __restrict__ b,
                               float* __restrict__ o, int vol) {
  int i = blockIdx.x * blockDim.x + threadIdx.x;
  if (i < vol) { o[i] = a[i]; o[vol + i] = b[i]; }
}

// fp32 OIDHW weights -> bf16 [CoutPad][Kpad] (K = Cin*27), zero padded
__global__ void repack_w_kernel(const float* __restrict__ w, bf16_t* __restrict__ wb,
                                int Cout, int K, int Kpad, int tot) {
  int i = blockIdx.x * blockDim.x + threadIdx.x;
  if (i >= tot) return;
  int co = i / Kpad;
  int k  = i - co * Kpad;
  float v = (co < Cout && k < K) ? w[(size_t)co * K + k] : 0.0f;
  wb[i] = (bf16_t)v;
}

// ---------------------------------------------------------------------------
// Implicit-GEMM conv3d (k=3, pad=1, stride=1), WMMA bf16 -> f32.
// Block: 256 threads = 8 waves; tile M=32 voxels x N=64 couts; K-step = 64.
// ---------------------------------------------------------------------------
__global__ __launch_bounds__(256) void conv3d_wmma_kernel(
    const float* __restrict__ in, const bf16_t* __restrict__ wb,
    const float* __restrict__ bias, float* __restrict__ out,
    int Cin, int Cout, int Dd, int Hh, int Ww, int Kpad) {
  __shared__ bf16_t As[32][72];   // 32 voxels x 64 K (+8 pad, keeps 16B align)
  __shared__ int    toff[27];     // tap -> input offset table

  const int tid  = threadIdx.x;
  const int lane = tid & 31;
  // wave id is uniform within the wave: pin it to an SGPR so all (ci,tap)
  // bookkeeping and the Cin guard become scalar (SALU + s_cbranch).
  const int wvu  = __builtin_amdgcn_readfirstlane(tid >> 5);
  const int mw   = wvu & 1;
  const int nw   = wvu >> 1;
  const int hw   = Hh * Ww;
  const int vol  = Dd * hw;

  // ---- one-time tap offset table ----
  if (tid < 27) {
    int kd = tid / 9, rr = tid - kd * 9;
    int kh = rr / 3, kw = rr - kh * 3;
    toff[tid] = (kd - 1) * hw + (kh - 1) * Ww + (kw - 1);
  }

  // ---- loader: lane = voxel (contiguous => coalesced); wave = 8 K-slots ----
  const int vxl = blockIdx.x * 32 + lane;
  const int d0  = vxl / hw;
  const int r0  = vxl - d0 * hw;
  const int h0  = r0 / Ww;
  const int w0  = r0 - h0 * Ww;

  // per-lane 27-bit halo validity mask (computed once)
  unsigned vmask = 0;
  for (int t = 0; t < 27; ++t) {
    int kd = t / 9, rr = t - kd * 9;
    int kh = rr / 3, kw = rr - kh * 3;
    int id = d0 + kd - 1, ih = h0 + kh - 1, iw = w0 + kw - 1;
    if ((unsigned)id < (unsigned)Dd && (unsigned)ih < (unsigned)Hh &&
        (unsigned)iw < (unsigned)Ww)
      vmask |= 1u << t;
  }

  // wave-uniform running (ci, tap) for K-slot run start; +64 per step
  const int kbase = wvu * 8;
  int ci_run = kbase / 27;
  int t_run  = kbase - ci_run * 27;

  // ---- fragment coords ----
  const int mrow  = mw * 16 + (lane & 15);
  const int khalf = lane >> 4;
  const int cout  = blockIdx.y * 64 + nw * 16 + (lane & 15);
  const bf16_t* wrow = wb + (size_t)cout * Kpad;

  v8f acc = {};

  for (int k0 = 0; k0 < Kpad; k0 += 64) {
    __syncthreads();  // previous fragment reads complete; toff visible
    // ---- im2col A-tile: 8 coalesced row-loads per wave, branchless halo ----
    {
      int ci = ci_run, t = t_run;                 // scalar
      const float* p = in + (size_t)ci * vol + vxl;
#pragma unroll
      for (int j = 0; j < 8; ++j) {
        bf16_t bv = (bf16_t)0.0f;
        int off = toff[t];                        // unconditional, uniform addr
        if (ci < Cin) {                           // scalar branch
          unsigned ok = (vmask >> t) & 1u;
          float v = p[ok ? off : 0];              // center tap always safe
          bv = (bf16_t)(ok ? v : 0.0f);
        }
        As[lane][kbase + j] = bv;
        if (++t == 27) { t = 0; ++ci; p += vol; }
      }
    }
    __syncthreads();
    // advance run start by 64 K (64 = 2*27 + 10)
    t_run += 10; ci_run += 2;
    if (t_run >= 27) { t_run -= 27; ++ci_run; }

    // ---- A fragments (ISA 16-bit 16x32: interleaved K-halves per lane) ----
    const bf16_t* pa0 = &As[mrow][khalf * 8];
    v8bf a0lo = *(const v8bf*)pa0;
    v8bf a0hi = *(const v8bf*)(pa0 + 16);
    const bf16_t* pa1 = pa0 + 32;
    v8bf a1lo = *(const v8bf*)pa1;
    v8bf a1hi = *(const v8bf*)(pa1 + 16);

    // ---- B fragments (same interleaved K-halves as A) ----
    const bf16_t* pb0 = wrow + k0 + khalf * 8;
    v8bf b0lo = *(const v8bf*)pb0;
    v8bf b0hi = *(const v8bf*)(pb0 + 16);
    const bf16_t* pb1 = pb0 + 32;
    v8bf b1lo = *(const v8bf*)pb1;
    v8bf b1hi = *(const v8bf*)(pb1 + 16);
    if (k0 + 128 < Kpad)
      __builtin_prefetch((const void*)(wrow + k0 + 128), 0, 1);

    v16bf a0, b0, a1, b1;
#pragma unroll
    for (int i = 0; i < 8; ++i) {
      a0[i] = a0lo[i]; a0[i + 8] = a0hi[i];
      b0[i] = b0lo[i]; b0[i + 8] = b0hi[i];
      a1[i] = a1lo[i]; a1[i + 8] = a1hi[i];
      b1[i] = b1lo[i]; b1[i + 8] = b1hi[i];
    }
    acc = __builtin_amdgcn_wmma_f32_16x16x32_bf16(false, a0, false, b0,
                                                  (short)0, acc, false, false);
    acc = __builtin_amdgcn_wmma_f32_16x16x32_bf16(false, a1, false, b1,
                                                  (short)0, acc, false, false);
  }

  // ---- epilogue: lane holds N=lane%16; M = r (+8 for hi lane half) ----
  if (cout < Cout) {
    float bv = bias[cout];
    int mbase = blockIdx.x * 32 + mw * 16 + khalf * 8;
    float* op = out + (size_t)cout * vol + mbase;
    float4 s0 = make_float4(acc[0] + bv, acc[1] + bv, acc[2] + bv, acc[3] + bv);
    float4 s1 = make_float4(acc[4] + bv, acc[5] + bv, acc[6] + bv, acc[7] + bv);
    *(float4*)op = s0;
    *((float4*)op + 1) = s1;
  }
}

// ---------------------------------------------------------------------------
// BatchNorm batch statistics: one block per channel
// ---------------------------------------------------------------------------
__global__ __launch_bounds__(256) void bn_stats_kernel(
    const float* __restrict__ x, int vol, int C, float* __restrict__ stats) {
  int c = blockIdx.x;
  const float* p = x + (size_t)c * vol;
  float s = 0.0f, s2 = 0.0f;
  for (int i = threadIdx.x; i < vol; i += 256) {
    float v = p[i]; s += v; s2 += v * v;
  }
  __shared__ float rs[256], rq[256];
  rs[threadIdx.x] = s; rq[threadIdx.x] = s2;
  __syncthreads();
  for (int st = 128; st > 0; st >>= 1) {
    if (threadIdx.x < st) {
      rs[threadIdx.x] += rs[threadIdx.x + st];
      rq[threadIdx.x] += rq[threadIdx.x + st];
    }
    __syncthreads();
  }
  if (threadIdx.x == 0) {
    float inv = 1.0f / (float)vol;
    float mu = rs[0] * inv;
    stats[c]     = mu;
    stats[C + c] = rq[0] * inv - mu * mu;
  }
}

__global__ void bn_apply_kernel(const float* __restrict__ x,
                                const float* __restrict__ stats,
                                const float* __restrict__ g,
                                const float* __restrict__ be,
                                int C, int vol, float* __restrict__ y) {
  int i = blockIdx.x * blockDim.x + threadIdx.x;
  if (i >= C * vol) return;
  int c = i / vol;
  float xn = (x[i] - stats[c]) * rsqrtf(stats[C + c] + BN_EPS);
  float o = g[c] * xn + be[c];
  y[i] = o > 0.0f ? o : 0.0f;
}

// ---------------------------------------------------------------------------
__global__ void maxpool_kernel(const float* __restrict__ x, float* __restrict__ y,
                               int C, int Dd, int Hh, int Ww) {
  int Do = Dd >> 1, Ho = Hh >> 1, Wo = Ww >> 1;
  int volo = Do * Ho * Wo;
  int i = blockIdx.x * blockDim.x + threadIdx.x;
  if (i >= C * volo) return;
  int c = i / volo;
  int v = i - c * volo;
  int od = v / (Ho * Wo);
  int r  = v - od * (Ho * Wo);
  int oh = r / Wo;
  int ow = r - oh * Wo;
  const float* p = x + (((size_t)c * Dd + od * 2) * Hh + oh * 2) * Ww + ow * 2;
  size_t sH = Ww, sD = (size_t)Hh * Ww;
  float m = p[0];
  m = fmaxf(m, p[1]);       m = fmaxf(m, p[sH]);      m = fmaxf(m, p[sH + 1]);
  m = fmaxf(m, p[sD]);      m = fmaxf(m, p[sD + 1]);
  m = fmaxf(m, p[sD + sH]); m = fmaxf(m, p[sD + sH + 1]);
  y[i] = m;
}

// ---------------------------------------------------------------------------
// ConvTranspose3d k=3, stride 2, pad(1,2) via lhs_dilation=2 correlation
// ---------------------------------------------------------------------------
__global__ void convt_kernel(const float* __restrict__ in, const float* __restrict__ w,
                             const float* __restrict__ bias, float* __restrict__ out,
                             int Cin, int Cout, int Dd, int Hh, int Ww) {
  int Do = Dd * 2, Ho = Hh * 2, Wo = Ww * 2;
  int volo = Do * Ho * Wo;
  int voli = Dd * Hh * Ww;
  int i = blockIdx.x * blockDim.x + threadIdx.x;
  if (i >= Cout * volo) return;
  int co = i / volo;
  int v  = i - co * volo;
  int od = v / (Ho * Wo);
  int r  = v - od * (Ho * Wo);
  int oh = r / Wo;
  int ow = r - oh * Wo;
  float acc = bias[co];
  const float* wbase = w + (size_t)co * Cin * 27;
  for (int kd = 0; kd < 3; ++kd) {
    int td = od + kd - 1;
    if (td < 0 || (td & 1) || (td >> 1) >= Dd) continue;
    int id = td >> 1;
    for (int kh = 0; kh < 3; ++kh) {
      int th = oh + kh - 1;
      if (th < 0 || (th & 1) || (th >> 1) >= Hh) continue;
      int ih = th >> 1;
      for (int kw = 0; kw < 3; ++kw) {
        int tw = ow + kw - 1;
        if (tw < 0 || (tw & 1) || (tw >> 1) >= Ww) continue;
        int iw = tw >> 1;
        const float* ip = in + ((size_t)id * Hh + ih) * Ww + iw;
        const float* wp = wbase + (kd * 9 + kh * 3 + kw);
        for (int ci = 0; ci < Cin; ++ci)
          acc += ip[(size_t)ci * voli] * wp[(size_t)ci * 27];
      }
    }
  }
  out[i] = acc;
}

// ---------------------------------------------------------------------------
// Trilinear warp (1-voxel zero pad + corner clamp, 8-corner gather)
// ---------------------------------------------------------------------------
__device__ __forceinline__ float warp_gather(const float* im, int Hh, int Ww, int Dd,
                                             int i, int j, int k) {
  i -= 1; j -= 1; k -= 1;
  if ((unsigned)i < (unsigned)Hh && (unsigned)j < (unsigned)Ww &&
      (unsigned)k < (unsigned)Dd)
    return im[((size_t)i * Ww + j) * Dd + k];
  return 0.0f;
}

__global__ void warp_kernel(const float* __restrict__ im, const float* __restrict__ flow,
                            float* __restrict__ out, int Hh, int Ww, int Dd) {
  int vol = Hh * Ww * Dd;
  int idx = blockIdx.x * blockDim.x + threadIdx.x;
  if (idx >= vol) return;
  int h = idx / (Ww * Dd);
  int r = idx - h * (Ww * Dd);
  int w = r / Dd;
  int d = r - w * Dd;
  float x = flow[idx]           + (float)h + 1.0f;
  float y = flow[vol + idx]     + (float)w + 1.0f;
  float z = flow[2 * vol + idx] + (float)d + 1.0f;
  float fx = floorf(x), fy = floorf(y), fz = floorf(z);
  float hiX = (float)(Hh + 1), hiY = (float)(Ww + 1), hiZ = (float)(Dd + 1);
  int x0 = (int)fminf(fmaxf(fx, 0.0f), hiX);
  int x1 = (int)fminf(fmaxf(fx + 1.0f, 0.0f), hiX);
  int y0 = (int)fminf(fmaxf(fy, 0.0f), hiY);
  int y1 = (int)fminf(fmaxf(fy + 1.0f, 0.0f), hiY);
  int z0 = (int)fminf(fmaxf(fz, 0.0f), hiZ);
  int z1 = (int)fminf(fmaxf(fz + 1.0f, 0.0f), hiZ);
  float dx = (float)x1 - x, dy = (float)y1 - y, dz = (float)z1 - z;
  float ex = 1.0f - dx, ey = 1.0f - dy, ez = 1.0f - dz;
  out[idx] = dx * dy * dz * warp_gather(im, Hh, Ww, Dd, x0, y0, z0)
           + dx * ey * dz * warp_gather(im, Hh, Ww, Dd, x0, y1, z0)
           + ex * dy * dz * warp_gather(im, Hh, Ww, Dd, x1, y0, z0)
           + ex * ey * dz * warp_gather(im, Hh, Ww, Dd, x1, y1, z0)
           + dx * dy * ez * warp_gather(im, Hh, Ww, Dd, x0, y0, z1)
           + dx * ey * ez * warp_gather(im, Hh, Ww, Dd, x0, y1, z1)
           + ex * dy * ez * warp_gather(im, Hh, Ww, Dd, x1, y0, z1)
           + ex * ey * ez * warp_gather(im, Hh, Ww, Dd, x1, y1, z1);
}

// ===========================================================================
// Host orchestration
// ===========================================================================
struct LayerP { const float *w, *b, *g, *be; };

extern "C" void kernel_launch(void* const* d_in, const int* in_sizes, int n_in,
                              void* d_out, int out_size, void* d_ws, size_t ws_size,
                              hipStream_t stream) {
  (void)in_sizes; (void)out_size; (void)ws_size;
  if (n_in < 70) return;  // 2 images + 17 layers * 4 tensors

  const int V96 = 96 * 96 * 96, V48 = 48 * 48 * 48, V24 = 24 * 24 * 24, V12 = 12 * 12 * 12;
  const float* moving = (const float*)d_in[0];
  const float* fixedI = (const float*)d_in[1];

  int pi = 2;
  auto nl = [&]() {
    LayerP L{(const float*)d_in[pi], (const float*)d_in[pi + 1],
             (const float*)d_in[pi + 2], (const float*)d_in[pi + 3]};
    pi += 4;
    return L;
  };
  LayerP e1a = nl(), e1b = nl();
  LayerP e2a = nl(), e2b = nl();
  LayerP e3a = nl(), e3b = nl();
  LayerP bt0 = nl(), bt1 = nl(), bt2 = nl();
  LayerP d3a = nl(), d3b = nl(), d3c = nl();
  LayerP d2a = nl(), d2b = nl(), d2c = nl();
  LayerP f0 = nl(), f1 = nl();

  // ---- workspace carving ----
  char* base = (char*)d_ws;
  size_t off = 0;
  auto af = [&](size_t n) -> float* {
    float* p = (float*)(base + off);
    off += ((n * sizeof(float) + 255) & ~(size_t)255);
    return p;
  };
  float* X  = af((size_t)2 * V96);
  float* CF = af((size_t)64 * V96);  float* e1 = CF + (size_t)32 * V96;
  float* C2 = af((size_t)128 * V48); float* e2 = C2 + (size_t)64 * V48;
  float* C3 = af((size_t)256 * V24); float* e3 = C3 + (size_t)128 * V24;
  float* P1 = af((size_t)32 * V48);
  float* P2 = af((size_t)64 * V24);
  float* P3 = af((size_t)128 * V12);
  float* TA = af((size_t)32 * V96);
  float* TB = af((size_t)32 * V96);
  float* FLOW  = af((size_t)3 * V96);
  float* STATS = af(1024);
  bf16_t* WBUF = (bf16_t*)(base + off);  off += (size_t)4 * 1024 * 1024;

  auto conv = [&](const float* in, int Cin, int Dd, int Hh, int Ww,
                  LayerP L, int Cout, float* out) {
    int K = Cin * 27;
    int Kpad  = (K + 63) & ~63;       // K-step is 64
    int CoutP = (Cout + 63) & ~63;
    int vol = Dd * Hh * Ww;
    int tot = CoutP * Kpad;
    repack_w_kernel<<<(tot + 255) / 256, 256, 0, stream>>>(L.w, WBUF, Cout, K, Kpad, tot);
    dim3 grid(vol / 32, CoutP / 64);
    conv3d_wmma_kernel<<<grid, 256, 0, stream>>>(in, WBUF, L.b, out,
                                                 Cin, Cout, Dd, Hh, Ww, Kpad);
  };
  auto bnrelu = [&](const float* x, LayerP L, int C, int vol, float* y) {
    bn_stats_kernel<<<C, 256, 0, stream>>>(x, vol, C, STATS);
    int tot = C * vol;
    bn_apply_kernel<<<(tot + 255) / 256, 256, 0, stream>>>(x, STATS, L.g, L.be, C, vol, y);
  };
  auto convT = [&](const float* in, int Cin, int Dd, int Hh, int Ww,
                   LayerP L, int Cout, float* out) {
    int tot = Cout * 8 * Dd * Hh * Ww;
    convt_kernel<<<(tot + 255) / 256, 256, 0, stream>>>(in, L.w, L.b, out,
                                                        Cin, Cout, Dd, Hh, Ww);
  };
  auto pool = [&](const float* x, float* y, int C, int Dd, int Hh, int Ww) {
    int tot = C * (Dd / 2) * (Hh / 2) * (Ww / 2);
    maxpool_kernel<<<(tot + 255) / 256, 256, 0, stream>>>(x, y, C, Dd, Hh, Ww);
  };

  concat2_kernel<<<(V96 + 255) / 256, 256, 0, stream>>>(moving, fixedI, X, V96);

  // encoder
  conv(X, 2, 96, 96, 96, e1a, 32, TA);      bnrelu(TA, e1a, 32, V96, TB);
  conv(TB, 32, 96, 96, 96, e1b, 32, TA);    bnrelu(TA, e1b, 32, V96, e1);
  pool(e1, P1, 32, 96, 96, 96);
  conv(P1, 32, 48, 48, 48, e2a, 64, TA);    bnrelu(TA, e2a, 64, V48, TB);
  conv(TB, 64, 48, 48, 48, e2b, 64, TA);    bnrelu(TA, e2b, 64, V48, e2);
  pool(e2, P2, 64, 48, 48, 48);
  conv(P2, 64, 24, 24, 24, e3a, 128, TA);   bnrelu(TA, e3a, 128, V24, TB);
  conv(TB, 128, 24, 24, 24, e3b, 128, TA);  bnrelu(TA, e3b, 128, V24, e3);
  pool(e3, P3, 128, 24, 24, 24);

  // bottleneck (last layer transposed, 12^3 -> 24^3) -> C3[0:128)
  conv(P3, 128, 12, 12, 12, bt0, 256, TA);  bnrelu(TA, bt0, 256, V12, TB);
  conv(TB, 256, 12, 12, 12, bt1, 128, TA);  bnrelu(TA, bt1, 128, V12, TB);
  convT(TB, 128, 12, 12, 12, bt2, 128, TA); bnrelu(TA, bt2, 128, V24, C3);

  // dec3 -> C2[0:64)
  conv(C3, 256, 24, 24, 24, d3a, 128, TA);  bnrelu(TA, d3a, 128, V24, TB);
  conv(TB, 128, 24, 24, 24, d3b, 128, TA);  bnrelu(TA, d3b, 128, V24, TB);
  convT(TB, 128, 24, 24, 24, d3c, 64, TA);  bnrelu(TA, d3c, 64, V48, C2);

  // dec2 -> CF[0:32)
  conv(C2, 128, 48, 48, 48, d2a, 64, TA);   bnrelu(TA, d2a, 64, V48, TB);
  conv(TB, 64, 48, 48, 48, d2b, 64, TA);    bnrelu(TA, d2b, 64, V48, TB);
  convT(TB, 64, 48, 48, 48, d2c, 32, TA);   bnrelu(TA, d2c, 32, V96, CF);

  // fin (64 -> 32 -> 3) -> FLOW
  conv(CF, 64, 96, 96, 96, f0, 32, TA);     bnrelu(TA, f0, 32, V96, TB);
  conv(TB, 32, 96, 96, 96, f1, 3, TA);      bnrelu(TA, f1, 3, V96, FLOW);

  // warp moving image by flow
  warp_kernel<<<(V96 + 255) / 256, 256, 0, stream>>>(moving, FLOW, (float*)d_out,
                                                     96, 96, 96);
}
